// GraphNetBlock_73684458930837
// MI455X (gfx1250) — matrix operand order repaired and checked by reading, not compiled, not run on hardware
//
#include <hip/hip_runtime.h>

typedef __bf16 bf16;
typedef __attribute__((ext_vector_type(8)))  float v8f;
typedef __attribute__((ext_vector_type(8)))  bf16  v8bf;
typedef __attribute__((ext_vector_type(16))) bf16  v16bf;

#define ZDIM 128
#define ROWS 128         // rows (edges/nodes) per block (4 waves x 32 rows)
#define SP   136         // LDS row stride in bf16 elements (128 + 8 pad)

// ---------------------------------------------------------------------------
// Stage ROWS rows x 128 f32 (gathered by ridx[]) into LDS as bf16.
// 128 threads, ROWS*32/128 float4 loads each.
// ---------------------------------------------------------------------------
__device__ __forceinline__ void stage_gather(bf16* __restrict__ X,
                                             const float* __restrict__ src,
                                             const int* __restrict__ ridx,
                                             int tid) {
#pragma unroll
  for (int i = 0; i < (ROWS * 32 / 128); ++i) {
    int flat = tid + (i << 7);           // 0..ROWS*32-1
    int row  = flat >> 5;                // 0..ROWS-1
    int c4   = flat & 31;                // float4 column
    const float4 v =
        reinterpret_cast<const float4*>(src)[(size_t)ridx[row] * (ZDIM / 4) + c4];
    union { bf16 h[4]; uint2 u; } pk;
    pk.h[0] = (bf16)v.x; pk.h[1] = (bf16)v.y;
    pk.h[2] = (bf16)v.z; pk.h[3] = (bf16)v.w;
    *reinterpret_cast<uint2*>(&X[row * SP + (c4 << 2)]) = pk.u;
  }
}

__device__ __forceinline__ void zero_acc(v8f acc[8]) {
#pragma unroll
  for (int nt = 0; nt < 8; ++nt)
#pragma unroll
    for (int r = 0; r < 8; ++r)
      acc[nt][r] = 0.f;
}

// ---------------------------------------------------------------------------
// One 32-row x 128-col WMMA pass over a 128-wide K chunk in LDS.
// Each B fragment feeds two WMMAs (two 16-row strips per wave).
// Wt is bf16 transposed weights, layout [N=128][K]; kofs selects the K chunk.
// ---------------------------------------------------------------------------
__device__ __forceinline__ void gemm_acc2(v8f acc0[8], v8f acc1[8],
                                          const bf16* __restrict__ X,
                                          const bf16* __restrict__ Wt, int K,
                                          int kofs, int wv, int lane) {
  const int hsel = lane >> 4;            // 0: lanes 0-15, 1: lanes 16-31
  const int l16  = lane & 15;
  const bf16* arow0 = X + (wv * 32 + l16) * SP + hsel * 16;
  const bf16* arow1 = arow0 + 16 * SP;
  const bf16* wl    = Wt + l16 * K + kofs + hsel * 16;
#pragma unroll
  for (int ks = 0; ks < 4; ++ks) {
    const int k0 = ks * 32;
    // A fragments: elems 0..15 = K(k0 + hsel*16 ..) contiguous
    v8bf a0lo = *reinterpret_cast<const v8bf*>(arow0 + k0);
    v8bf a0hi = *reinterpret_cast<const v8bf*>(arow0 + k0 + 8);
    v8bf a1lo = *reinterpret_cast<const v8bf*>(arow1 + k0);
    v8bf a1hi = *reinterpret_cast<const v8bf*>(arow1 + k0 + 8);
    v16bf a0 = __builtin_shufflevector(a0lo, a0hi, 0, 1, 2, 3, 4, 5, 6, 7,
                                       8, 9, 10, 11, 12, 13, 14, 15);
    v16bf a1 = __builtin_shufflevector(a1lo, a1hi, 0, 1, 2, 3, 4, 5, 6, 7,
                                       8, 9, 10, 11, 12, 13, 14, 15);
#pragma unroll
    for (int nt = 0; nt < 8; ++nt) {
      // B fragment: 16 contiguous bf16 of column n = nt*16 + l16
      v16bf b = *reinterpret_cast<const v16bf*>(wl + nt * 16 * K + k0);
      acc0[nt] = __builtin_amdgcn_wmma_f32_16x16x32_bf16(
          false, a0, false, b, (short)0, acc0[nt], false, false);
      acc1[nt] = __builtin_amdgcn_wmma_f32_16x16x32_bf16(
          false, a1, false, b, (short)0, acc1[nt], false, false);
    }
  }
}

// bias + ReLU + write bf16 activations back to LDS for the next layer
__device__ __forceinline__ void epi_relu_lds(const v8f acc0[8],
                                             const v8f acc1[8],
                                             const float* __restrict__ bias,
                                             bf16* __restrict__ X, int wv,
                                             int lane) {
  const int hsel = lane >> 4, l16 = lane & 15;
#pragma unroll
  for (int nt = 0; nt < 8; ++nt) {
    float bv = bias[nt * 16 + l16];
#pragma unroll
    for (int r = 0; r < 8; ++r) {
      float v0 = acc0[nt][r] + bv;
      float v1 = acc1[nt][r] + bv;
      v0 = v0 > 0.f ? v0 : 0.f;
      v1 = v1 > 0.f ? v1 : 0.f;
      int rr = hsel * 8 + r;
      X[(wv * 32 + rr) * SP + nt * 16 + l16]      = (bf16)v0;
      X[(wv * 32 + 16 + rr) * SP + nt * 16 + l16] = (bf16)v1;
    }
  }
}

// ---------------------------------------------------------------------------
// Edge MLP: e_upd = MLP([nf[s], nf[r], ef]); out_e = e_upd + ef;
// atomicAdd e_upd into agg[receiver].
// ---------------------------------------------------------------------------
__global__ void __launch_bounds__(128)
gnb_edge_kernel(const float* __restrict__ nf, const float* __restrict__ ef,
                const int* __restrict__ snd, const int* __restrict__ rcv,
                const bf16* __restrict__ We1t, const float* __restrict__ be1,
                const bf16* __restrict__ We2t, const float* __restrict__ be2,
                const bf16* __restrict__ We3t, const float* __restrict__ be3,
                float* agg, float* __restrict__ out_e, int E) {
  __shared__ __align__(16) bf16 X[ROWS * SP];
  __shared__ int iS[ROWS], iR[ROWS], iE[ROWS];
  const int tid = threadIdx.x;
  const int lane = tid & 31, wv = tid >> 5;
  const int e0 = blockIdx.x * ROWS;

  {
    int e = e0 + tid;
    if (e >= E) e = E - 1;
    iS[tid] = snd[e]; iR[tid] = rcv[e]; iE[tid] = e;
  }
  __syncthreads();

  v8f acc0[8], acc1[8];
  zero_acc(acc0); zero_acc(acc1);

  // layer 1: K = 384 in three 128-chunks (sender, receiver, edge feats)
  stage_gather(X, nf, iS, tid);
  __syncthreads();
  gemm_acc2(acc0, acc1, X, We1t, 384, 0, wv, lane);
  __syncthreads();
  stage_gather(X, nf, iR, tid);
  __syncthreads();
  gemm_acc2(acc0, acc1, X, We1t, 384, 128, wv, lane);
  __syncthreads();
  stage_gather(X, ef, iE, tid);
  __syncthreads();
  gemm_acc2(acc0, acc1, X, We1t, 384, 256, wv, lane);
  __syncthreads();
  epi_relu_lds(acc0, acc1, be1, X, wv, lane);
  __syncthreads();

  // layer 2
  zero_acc(acc0); zero_acc(acc1);
  gemm_acc2(acc0, acc1, X, We2t, 128, 0, wv, lane);
  __syncthreads();
  epi_relu_lds(acc0, acc1, be2, X, wv, lane);
  __syncthreads();

  // layer 3 (no relu) + residual + scatter-add
  zero_acc(acc0); zero_acc(acc1);
  gemm_acc2(acc0, acc1, X, We3t, 128, 0, wv, lane);

  const int hsel = lane >> 4, l16 = lane & 15;
#pragma unroll
  for (int nt = 0; nt < 8; ++nt) {
    float bv = be3[nt * 16 + l16];
#pragma unroll
    for (int r = 0; r < 8; ++r) {
#pragma unroll
      for (int s = 0; s < 2; ++s) {
        int lrow = wv * 32 + s * 16 + hsel * 8 + r;
        int ge = e0 + lrow;
        if (ge < E) {
          float v = (s == 0 ? acc0[nt][r] : acc1[nt][r]) + bv;
          int col = nt * 16 + l16;
          size_t o = (size_t)ge * ZDIM + col;
          out_e[o] = v + ef[o];
          atomicAdd(&agg[(size_t)iR[lrow] * ZDIM + col], v);
        }
      }
    }
  }
}

// ---------------------------------------------------------------------------
// Node MLP: n_upd = MLP([nf, agg]); out_n = n_upd + nf.  (agg aliases out_n)
// ---------------------------------------------------------------------------
__global__ void __launch_bounds__(128)
gnb_node_kernel(const float* __restrict__ nf, const float* agg,
                const bf16* __restrict__ Wn1t, const float* __restrict__ bn1,
                const bf16* __restrict__ Wn2t, const float* __restrict__ bn2,
                const bf16* __restrict__ Wn3t, const float* __restrict__ bn3,
                float* out_n, int N) {
  __shared__ __align__(16) bf16 X[ROWS * SP];
  __shared__ int iN[ROWS];
  const int tid = threadIdx.x;
  const int lane = tid & 31, wv = tid >> 5;
  const int n0 = blockIdx.x * ROWS;

  {
    int n = n0 + tid;
    if (n >= N) n = N - 1;
    iN[tid] = n;
  }
  __syncthreads();

  v8f acc0[8], acc1[8];
  zero_acc(acc0); zero_acc(acc1);

  // layer 1: K = 256 in two 128-chunks (node feats, aggregated messages)
  stage_gather(X, nf, iN, tid);
  __syncthreads();
  gemm_acc2(acc0, acc1, X, Wn1t, 256, 0, wv, lane);
  __syncthreads();
  stage_gather(X, agg, iN, tid);
  __syncthreads();
  gemm_acc2(acc0, acc1, X, Wn1t, 256, 128, wv, lane);
  __syncthreads();
  epi_relu_lds(acc0, acc1, bn1, X, wv, lane);
  __syncthreads();

  // layer 2
  zero_acc(acc0); zero_acc(acc1);
  gemm_acc2(acc0, acc1, X, Wn2t, 128, 0, wv, lane);
  __syncthreads();
  epi_relu_lds(acc0, acc1, bn2, X, wv, lane);
  __syncthreads();

  // layer 3 + residual
  zero_acc(acc0); zero_acc(acc1);
  gemm_acc2(acc0, acc1, X, Wn3t, 128, 0, wv, lane);

  const int hsel = lane >> 4, l16 = lane & 15;
#pragma unroll
  for (int nt = 0; nt < 8; ++nt) {
    float bv = bn3[nt * 16 + l16];
#pragma unroll
    for (int r = 0; r < 8; ++r) {
#pragma unroll
      for (int s = 0; s < 2; ++s) {
        int lrow = wv * 32 + s * 16 + hsel * 8 + r;
        int gn = n0 + lrow;
        if (gn < N) {
          float v = (s == 0 ? acc0[nt][r] : acc1[nt][r]) + bv;
          int col = nt * 16 + l16;
          size_t o = (size_t)gn * ZDIM + col;
          out_n[o] = v + nf[o];
        }
      }
    }
  }
}

// ---------------------------------------------------------------------------
// Weight prep: f32 [K][128] row-major  ->  bf16 [128][K] (transposed).
// ---------------------------------------------------------------------------
__global__ void __launch_bounds__(256)
wprep_kernel(const float* __restrict__ src, bf16* __restrict__ dst, int K) {
  int i = blockIdx.x * 256 + threadIdx.x;   // i = k*128 + n
  if (i < K * 128) {
    int k = i >> 7, n = i & 127;
    dst[n * K + k] = (bf16)src[i];
  }
}

extern "C" void kernel_launch(void* const* d_in, const int* in_sizes, int n_in,
                              void* d_out, int out_size, void* d_ws,
                              size_t ws_size, hipStream_t stream) {
  const float* nf  = (const float*)d_in[0];
  const float* ef  = (const float*)d_in[1];
  const int*   snd = (const int*)d_in[2];
  const int*   rcv = (const int*)d_in[3];
  const float* We1 = (const float*)d_in[4];  const float* be1 = (const float*)d_in[5];
  const float* We2 = (const float*)d_in[6];  const float* be2 = (const float*)d_in[7];
  const float* We3 = (const float*)d_in[8];  const float* be3 = (const float*)d_in[9];
  const float* Wn1 = (const float*)d_in[10]; const float* bn1 = (const float*)d_in[11];
  const float* Wn2 = (const float*)d_in[12]; const float* bn2 = (const float*)d_in[13];
  const float* Wn3 = (const float*)d_in[14]; const float* bn3 = (const float*)d_in[15];

  const int N = in_sizes[0] / ZDIM;   // 100000
  const int E = in_sizes[1] / ZDIM;   // 600000

  float* out_n = (float*)d_out;                      // also agg scratch
  float* out_e = out_n + (size_t)N * ZDIM;

  bf16* ws   = (bf16*)d_ws;
  bf16* We1t = ws;                                   // 384*128
  bf16* We2t = We1t + 384 * 128;                     // 128*128
  bf16* We3t = We2t + 128 * 128;                     // 128*128
  bf16* Wn1t = We3t + 128 * 128;                     // 256*128
  bf16* Wn2t = Wn1t + 256 * 128;                     // 128*128
  bf16* Wn3t = Wn2t + 128 * 128;                     // 128*128

  wprep_kernel<<<(384 * 128 + 255) / 256, 256, 0, stream>>>(We1, We1t, 384);
  wprep_kernel<<<(128 * 128 + 255) / 256, 256, 0, stream>>>(We2, We2t, 128);
  wprep_kernel<<<(128 * 128 + 255) / 256, 256, 0, stream>>>(We3, We3t, 128);
  wprep_kernel<<<(256 * 128 + 255) / 256, 256, 0, stream>>>(Wn1, Wn1t, 256);
  wprep_kernel<<<(128 * 128 + 255) / 256, 256, 0, stream>>>(Wn2, Wn2t, 128);
  wprep_kernel<<<(128 * 128 + 255) / 256, 256, 0, stream>>>(Wn3, Wn3t, 128);

  // zero agg (node section of d_out); edge section is fully overwritten
  hipMemsetAsync(out_n, 0, (size_t)N * ZDIM * sizeof(float), stream);

  gnb_edge_kernel<<<(E + ROWS - 1) / ROWS, 128, 0, stream>>>(
      nf, ef, snd, rcv, We1t, be1, We2t, be2, We3t, be3, out_n, out_e, E);

  gnb_node_kernel<<<(N + ROWS - 1) / ROWS, 128, 0, stream>>>(
      nf, out_n, Wn1t, bn1, Wn2t, bn2, Wn3t, bn3, out_n, N);
}
